// InBatchNegativeInfoNCELoss_81088982548553
// MI455X (gfx1250) — matrix operand order repaired and checked by reading, not compiled, not run on hardware
//
#include <hip/hip_runtime.h>

#define N_EMB 8192
#define D_EMB 256
#define INV_T 20.0f   // 1 / 0.05
#define EPS_N 1e-8f

typedef __attribute__((ext_vector_type(16))) __bf16  v16bf;
typedef __attribute__((ext_vector_type(8)))  float   v8f;
typedef __attribute__((ext_vector_type(4)))  unsigned int u32x4;

union BF16Frag { v16bf v; u32x4 u[2]; };

__device__ inline unsigned short f32_to_bf16(float f) {
  unsigned u = __builtin_bit_cast(unsigned, f);
  u += 0x7FFFu + ((u >> 16) & 1u);        // round-to-nearest-even
  return (unsigned short)(u >> 16);
}

// ---------------------------------------------------------------------------
// Kernel 0: zero the row/col exp-sum accumulators.
// ---------------------------------------------------------------------------
__global__ void zero_kernel(float* __restrict__ buf, int n) {
  int i = blockIdx.x * blockDim.x + threadIdx.x;
  if (i < n) buf[i] = 0.0f;
}

// ---------------------------------------------------------------------------
// Kernel 1: L2-normalize each row (D=256, one block per row) and store bf16.
// Blocks [0, N) handle q, blocks [N, 2N) handle p.
// ---------------------------------------------------------------------------
__global__ __launch_bounds__(256)
void normalize_bf16_kernel(const float* __restrict__ q,
                           const float* __restrict__ p,
                           unsigned short* __restrict__ qn,
                           unsigned short* __restrict__ pn) {
  int r = blockIdx.x;
  const float* src;
  unsigned short* dst;
  if (r < N_EMB) { src = q + (size_t)r * D_EMB;           dst = qn + (size_t)r * D_EMB; }
  else           { src = p + (size_t)(r - N_EMB) * D_EMB; dst = pn + (size_t)(r - N_EMB) * D_EMB; }

  int t = threadIdx.x;                     // 0..255, D == blockDim
  float x = src[t];
  float v = x * x;
  // wave32 reduction
  #pragma unroll
  for (int m = 1; m < 32; m <<= 1) v += __shfl_xor(v, m, 32);

  __shared__ float smem[8];
  if ((t & 31) == 0) smem[t >> 5] = v;
  __syncthreads();
  float ss = 0.0f;
  #pragma unroll
  for (int i = 0; i < 8; ++i) ss += smem[i];

  float scale = 1.0f / fmaxf(sqrtf(ss), EPS_N);
  dst[t] = f32_to_bf16(x * scale);
}

// ---------------------------------------------------------------------------
// Kernel 2: fused bf16 WMMA similarity tiles + exp + row/col sum + diagonal.
// Block = 256 threads = 8 waves. Block tile = 128 rows x 64 cols.
// Wave w computes rows [rowBase + 16w, +16) x all 64 cols (4 WMMA accumulators).
// ---------------------------------------------------------------------------
__global__ __launch_bounds__(256)
void sim_exp_kernel(const unsigned short* __restrict__ qn,
                    const unsigned short* __restrict__ pn,
                    float* __restrict__ row_sum,
                    float* __restrict__ col_sum,
                    float* __restrict__ diag) {
  const int lane = threadIdx.x & 31;
  const int wave = threadIdx.x >> 5;        // 0..7
  const int half = lane >> 4;               // 0/1
  const int l15  = lane & 15;

  const int colBase = blockIdx.x * 64;
  const int rowBase = blockIdx.y * 128 + wave * 16;

  v8f acc[4];
  #pragma unroll
  for (int t = 0; t < 4; ++t) acc[t] = (v8f){};

  // A fragment: lane holds row (l&15); K chunks at 8*half and 16+8*half.
  const unsigned short* arow = qn + (size_t)(rowBase + l15) * D_EMB;

  #pragma unroll
  for (int kt = 0; kt < D_EMB; kt += 32) {
    BF16Frag a;
    a.u[0] = *(const u32x4*)(arow + kt + 8 * half);
    a.u[1] = *(const u32x4*)(arow + kt + 16 + 8 * half);

    #pragma unroll
    for (int t = 0; t < 4; ++t) {
      // B fragment: lane holds column (l&15) of P^T == row of P;
      // K chunks at kt + 8*half and kt + 16 + 8*half (same as A).
      const unsigned short* bp =
          pn + (size_t)(colBase + t * 16 + l15) * D_EMB + kt + 8 * half;
      BF16Frag b;
      b.u[0] = *(const u32x4*)(bp);
      b.u[1] = *(const u32x4*)(bp + 16);

      acc[t] = __builtin_amdgcn_wmma_f32_16x16x32_bf16(
          false, a.v, false, b.v, (short)0, acc[t], false, false);
    }
  }

  // Epilogue: scores = acc * INV_T; accumulate exp() into row/col sums.
  // C/D layout: lane -> column n = l&15 ; VGPR v -> row m = v + 8*half.
  float rowpart[8];
  #pragma unroll
  for (int v = 0; v < 8; ++v) rowpart[v] = 0.0f;

  #pragma unroll
  for (int t = 0; t < 4; ++t) {
    float cp = 0.0f;
    #pragma unroll
    for (int v = 0; v < 8; ++v) {
      float s  = acc[t][v] * INV_T;
      float ex = __expf(s);
      rowpart[v] += ex;
      cp += ex;
      int gr = rowBase + v + 8 * half;
      int gc = colBase + t * 16 + l15;
      if (gr == gc) diag[gr] = s;           // each diag element written once
    }
    cp += __shfl_xor(cp, 16, 32);           // combine the two row-halves
    if (half == 0)
      atomicAdd(&col_sum[colBase + t * 16 + l15], cp);
  }

  #pragma unroll
  for (int v = 0; v < 8; ++v) {
    float rv = rowpart[v];
    rv += __shfl_xor(rv, 1, 32);
    rv += __shfl_xor(rv, 2, 32);
    rv += __shfl_xor(rv, 4, 32);
    rv += __shfl_xor(rv, 8, 32);            // sum over 16 columns in this half
    if (l15 == 0)
      atomicAdd(&row_sum[rowBase + v + 8 * half], rv);
  }
}

// ---------------------------------------------------------------------------
// Kernel 3: loss = mean(log(row_sum) - diag) + mean(log(col_sum) - diag)
// ---------------------------------------------------------------------------
__global__ __launch_bounds__(256)
void finalize_kernel(const float* __restrict__ row_sum,
                     const float* __restrict__ col_sum,
                     const float* __restrict__ diag,
                     float* __restrict__ out) {
  float acc = 0.0f;
  for (int i = threadIdx.x; i < N_EMB; i += 256)
    acc += logf(row_sum[i]) + logf(col_sum[i]) - 2.0f * diag[i];

  #pragma unroll
  for (int m = 1; m < 32; m <<= 1) acc += __shfl_xor(acc, m, 32);

  __shared__ float smem[8];
  if ((threadIdx.x & 31) == 0) smem[threadIdx.x >> 5] = acc;
  __syncthreads();
  if (threadIdx.x == 0) {
    float s = 0.0f;
    #pragma unroll
    for (int i = 0; i < 8; ++i) s += smem[i];
    out[0] = s / (float)N_EMB;
  }
}

// ---------------------------------------------------------------------------
extern "C" void kernel_launch(void* const* d_in, const int* in_sizes, int n_in,
                              void* d_out, int out_size, void* d_ws, size_t ws_size,
                              hipStream_t stream) {
  const float* q = (const float*)d_in[0];
  const float* p = (const float*)d_in[1];

  // Workspace layout: qn bf16 [N*D] | pn bf16 [N*D] | row_sum | col_sum | diag
  unsigned short* qn = (unsigned short*)d_ws;
  unsigned short* pn = qn + (size_t)N_EMB * D_EMB;
  float* row_sum = (float*)(pn + (size_t)N_EMB * D_EMB);
  float* col_sum = row_sum + N_EMB;
  float* diag    = col_sum + N_EMB;
  float* out     = (float*)d_out;

  zero_kernel<<<(2 * N_EMB + 255) / 256, 256, 0, stream>>>(row_sum, 2 * N_EMB);
  normalize_bf16_kernel<<<2 * N_EMB, 256, 0, stream>>>(q, p, qn, pn);

  dim3 grid(N_EMB / 64, N_EMB / 128);   // 128 x 64 tiles
  sim_exp_kernel<<<grid, 256, 0, stream>>>(qn, pn, row_sum, col_sum, diag);

  finalize_kernel<<<1, 256, 0, stream>>>(row_sum, col_sum, diag, out);
}
